// generator_13950053778302
// MI455X (gfx1250) — matrix operand order repaired and checked by reading, not compiled, not run on hardware
//
#include <hip/hip_runtime.h>
#include <hip/hip_bf16.h>

// ---------------------------------------------------------------------------
// Types for WMMA (gfx1250, wave32)
// ---------------------------------------------------------------------------
typedef __attribute__((ext_vector_type(16))) __bf16       v16bf;
typedef __attribute__((ext_vector_type(8)))  float        v8f;
typedef __attribute__((ext_vector_type(4)))  unsigned int u32x4;

union V16U { u32x4 u[2]; v16bf v; };

// Problem constants
#define B_    32
#define S_    128
#define V_    32000
#define E_    512
#define G_    1024
#define HI_   512
#define NN_   16
#define NH_   8
#define DH_   64
#define LOGITS_ELEMS ((size_t)B_ * 127 * V_)   // 130,048,000

// ---------------------------------------------------------------------------
// Helpers
// ---------------------------------------------------------------------------
__device__ __forceinline__ unsigned short f2bf_u(float f) {
  unsigned u = __float_as_uint(f);
  unsigned r = (u + 0x7FFFu + ((u >> 16) & 1u)) >> 16;   // round-to-nearest-even
  return (unsigned short)r;
}
__device__ __forceinline__ float sigf(float x)  { return 1.0f / (1.0f + __expf(-x)); }
__device__ __forceinline__ float tanh_f(float x){ return 2.0f / (1.0f + __expf(-2.0f * x)) - 1.0f; }

// A-fragment (16x32 bf16, M x K). Lane L holds row M=L%16.
// ISA 7.12.2: lanes 0-15 -> K chunks [0..7] & [16..23]; lanes 16-31 -> [8..15] & [24..31].
__device__ __forceinline__ v16bf load_a16(const unsigned short* base, int ld,
                                          int row, int k, int lane) {
  int klo = (lane & 16) ? 8 : 0;
  const unsigned short* p = base + (size_t)row * ld + k + klo;
  V16U t;
  t.u[0] = *(const u32x4*)(p);        // 8 bf16: K = k+klo .. k+klo+7
  t.u[1] = *(const u32x4*)(p + 16);   // 8 bf16: K = k+klo+16 .. k+klo+23
  return t.v;
}
// B-fragment (32x16 bf16, K x N) from row-major W[N][K] (we compute X @ W^T).
// Lane L holds column N=L%16; lanes 0-15 -> K [0..7] & [16..23]; lanes 16-31 -> [8..15] & [24..31].
__device__ __forceinline__ v16bf load_b16(const unsigned short* base, int ld,
                                          int col, int k, int lane) {
  int klo = (lane & 16) ? 8 : 0;
  const unsigned short* p = base + (size_t)col * ld + k + klo;
  V16U t;
  t.u[0] = *(const u32x4*)(p);
  t.u[1] = *(const u32x4*)(p + 16);
  return t.v;
}
__device__ __forceinline__ v8f wmma_bf16(v16bf a, v16bf b, v8f c) {
  return __builtin_amdgcn_wmma_f32_16x16x32_bf16(false, a, false, b, (short)0, c,
                                                 false, false);
}

// ---------------------------------------------------------------------------
// fp32 -> bf16 weight conversion
// ---------------------------------------------------------------------------
__global__ void cvt_bf16(const float* __restrict__ src, unsigned short* __restrict__ dst, int n) {
  int i = blockIdx.x * blockDim.x + threadIdx.x;
  if (i < n) dst[i] = f2bf_u(src[i]);
}

// ---------------------------------------------------------------------------
// Fused MHA + concat + img_embed = tanh(ff @ W_img^T + b_img)  (one block per batch)
// Writes bf16 img_embed into X row t=0.
// ---------------------------------------------------------------------------
__global__ __launch_bounds__(256) void attn_kernel(
    const float* __restrict__ entity, const float* __restrict__ neigh,
    const int* __restrict__ nnum, const float* __restrict__ imgf,
    const float* __restrict__ Wq, const float* __restrict__ bq,
    const float* __restrict__ Wk, const float* __restrict__ bk,
    const float* __restrict__ Wv, const float* __restrict__ bv,
    const float* __restrict__ Wo, const float* __restrict__ bo,
    const float* __restrict__ W_img, const float* __restrict__ b_img,
    float* __restrict__ kbuf, float* __restrict__ vbuf,
    unsigned short* __restrict__ Xbf) {
  __shared__ float sN[NN_ * HI_];  // neighbors
  __shared__ float sE[HI_];        // entity
  __shared__ float sI[HI_];        // img feature
  __shared__ float sQ[HI_];        // q
  __shared__ float sO[HI_];        // attn @ v
  __shared__ float sP[HI_];        // attn_out projected
  __shared__ float sS[NH_ * NN_];  // scores / softmax

  const int b   = blockIdx.x;
  const int tid = threadIdx.x;

  for (int j = tid; j < HI_; j += 256) { sE[j] = entity[b * HI_ + j]; sI[j] = imgf[b * HI_ + j]; }
  for (int j = tid; j < NN_ * HI_; j += 256) sN[j] = neigh[(size_t)b * NN_ * HI_ + j];
  __syncthreads();

  // q = entity @ Wq^T + bq
  for (int j = tid; j < HI_; j += 256) {
    float acc = bq[j];
    const float* w = Wq + (size_t)j * HI_;
    for (int e = 0; e < HI_; ++e) acc += sE[e] * w[e];
    sQ[j] = acc;
  }
  // k,v = neigh @ Wk^T/Wv^T (to global scratch)
  for (int idx = tid; idx < NN_ * HI_; idx += 256) {
    int n = idx >> 9, j = idx & (HI_ - 1);
    float aK = bk[j], aV = bv[j];
    const float* wk = Wk + (size_t)j * HI_;
    const float* wv = Wv + (size_t)j * HI_;
    const float* nb = sN + n * HI_;
    for (int e = 0; e < HI_; ++e) { float f = nb[e]; aK += f * wk[e]; aV += f * wv[e]; }
    kbuf[(size_t)b * NN_ * HI_ + idx] = aK;
    vbuf[(size_t)b * NN_ * HI_ + idx] = aV;
  }
  __syncthreads();

  // scores (8 heads x 16 keys), scale 1/sqrt(64), mask n >= nnum[b]
  if (tid < NH_ * NN_) {
    int h = tid >> 4, n = tid & 15;
    float s = 0.0f;
    const float* kk = kbuf + (size_t)b * NN_ * HI_ + n * HI_ + h * DH_;
    for (int d = 0; d < DH_; ++d) s += sQ[h * DH_ + d] * kk[d];
    s *= 0.125f;
    if (n >= nnum[b]) s = -1000000000.0f;
    sS[tid] = s;
  }
  __syncthreads();
  if (tid < NH_) {
    float mx = -3.4e38f;
    for (int n = 0; n < NN_; ++n) mx = fmaxf(mx, sS[tid * NN_ + n]);
    float sum = 0.0f;
    for (int n = 0; n < NN_; ++n) { float e = __expf(sS[tid * NN_ + n] - mx); sS[tid * NN_ + n] = e; sum += e; }
    float inv = 1.0f / sum;
    for (int n = 0; n < NN_; ++n) sS[tid * NN_ + n] *= inv;
  }
  __syncthreads();

  // out = attn @ v
  for (int j = tid; j < HI_; j += 256) {
    int h = j >> 6;
    float o = 0.0f;
    const float* vv = vbuf + (size_t)b * NN_ * HI_ + j;
    for (int n = 0; n < NN_; ++n) o += sS[h * NN_ + n] * vv[n * HI_];
    sO[j] = o;
  }
  __syncthreads();
  // attn_out = out @ Wo^T + bo
  for (int j = tid; j < HI_; j += 256) {
    float acc = bo[j];
    const float* w = Wo + (size_t)j * HI_;
    for (int e = 0; e < HI_; ++e) acc += sO[e] * w[e];
    sP[j] = acc;
  }
  __syncthreads();
  // img_embed = tanh([entity, attn_out, img] @ W_img^T + b_img) -> X row t=0 (bf16)
  for (int j = tid; j < E_; j += 256) {
    float acc = b_img[j];
    const float* w = W_img + (size_t)j * (3 * HI_);
    for (int e = 0; e < HI_; ++e) acc += sE[e] * w[e];
    for (int e = 0; e < HI_; ++e) acc += sP[e] * w[HI_ + e];
    for (int e = 0; e < HI_; ++e) acc += sI[e] * w[2 * HI_ + e];
    Xbf[(size_t)b * E_ + j] = f2bf_u(tanh_f(acc));
  }
}

// ---------------------------------------------------------------------------
// Embedding gather -> bf16 X rows t=1..127
// ---------------------------------------------------------------------------
__global__ __launch_bounds__(256) void gather_embed(
    const int* __restrict__ sent, const float* __restrict__ emb,
    unsigned short* __restrict__ Xbf) {
  int r  = blockIdx.x;           // 0..4063
  int tx = (r >> 5) + 1;         // 1..127
  int b  = r & 31;
  int word = sent[b * S_ + (tx - 1)];
  const float* e = emb + (size_t)word * E_;
  unsigned short* d = Xbf + ((size_t)tx * B_ + b) * E_;
  for (int j = threadIdx.x; j < E_; j += 256) d[j] = f2bf_u(e[j]);
}

// ---------------------------------------------------------------------------
// One LSTM step (launched 128x). 128 waves; each wave: 16(M) x 16(N) gate tile
// across all four gate sections (A reused 4x), elementwise finished in-wave.
// ---------------------------------------------------------------------------
__global__ __launch_bounds__(128) void lstm_step_kernel(
    const unsigned short* __restrict__ Xbf, unsigned short* __restrict__ Hbf,
    const unsigned short* __restrict__ Wih, const unsigned short* __restrict__ Whh,
    const float* __restrict__ b_ih, const float* __restrict__ b_hh,
    float* __restrict__ Ccar, int t) {
  const int gtid = blockIdx.x * 128 + threadIdx.x;
  const int w    = gtid >> 5;          // 0..127
  const int lane = threadIdx.x & 31;
  const int mt   = w >> 6;             // 0..1 (batch tile)
  const int j0   = (w & 63) * 16;      // gate column base within [0,1024)
  const int rowA = mt * 16 + (lane & 15);
  const int nloc = lane & 15;
  const int half = (lane >> 4) & 1;

  v8f acc[4] = {};

  const unsigned short* Xt = Xbf + (size_t)t * B_ * E_;
  for (int k = 0; k < E_; k += 32) {
    v16bf a = load_a16(Xt, E_, rowA, k, lane);
#pragma unroll
    for (int gi = 0; gi < 4; ++gi) {
      v16bf bb = load_b16(Wih, E_, gi * G_ + j0 + nloc, k, lane);
      acc[gi] = wmma_bf16(a, bb, acc[gi]);
    }
  }
  if (t > 0) {
    const unsigned short* Hp = Hbf + (size_t)(t - 1) * B_ * G_;
    for (int k = 0; k < G_; k += 32) {
      v16bf a = load_a16(Hp, G_, rowA, k, lane);
#pragma unroll
      for (int gi = 0; gi < 4; ++gi) {
        v16bf bb = load_b16(Whh, G_, gi * G_ + j0 + nloc, k, lane);
        acc[gi] = wmma_bf16(a, bb, acc[gi]);
      }
    }
  }

  const int g  = j0 + nloc;
  const float bI = b_ih[g]            + b_hh[g];
  const float bF = b_ih[G_ + g]       + b_hh[G_ + g];
  const float bG = b_ih[2 * G_ + g]   + b_hh[2 * G_ + g];
  const float bO = b_ih[3 * G_ + g]   + b_hh[3 * G_ + g];

#pragma unroll
  for (int r = 0; r < 8; ++r) {
    int bb = mt * 16 + r + half * 8;                 // batch index
    float gi_ = acc[0][r] + bI;
    float gf_ = acc[1][r] + bF;
    float gg_ = acc[2][r] + bG;
    float go_ = acc[3][r] + bO;
    float c_old = (t > 0) ? Ccar[bb * G_ + g] : 0.0f;
    float c_new = sigf(gf_) * c_old + sigf(gi_) * tanh_f(gg_);
    Ccar[bb * G_ + g] = c_new;
    float h = sigf(go_) * tanh_f(c_new);
    Hbf[(size_t)t * B_ * G_ + (size_t)bb * G_ + g] = f2bf_u(h);
  }
}

// ---------------------------------------------------------------------------
// logits = H[1..127] @ W_out^T + b_out   (M=4064, N=32000, K=1024)
// Block: 8 waves (4m x 2n), LDS double-buffered 64(M) x 128(N) tile, KC=64:
// each A/B element enters the block exactly once from L2 (~6.3 GB total), and
// each wave runs 8 WMMAs (2 K-substeps x 4 n-tiles) per __syncthreads.
// 48 KB LDS -> 6 blocks per 320 KB WGP.
// ---------------------------------------------------------------------------
#define KC 64   // K chunk

__global__ __launch_bounds__(256) void out_gemm(
    const unsigned short* __restrict__ Hbf, const unsigned short* __restrict__ Wout,
    const float* __restrict__ b_out, float* __restrict__ out) {
  __shared__ __attribute__((aligned(16))) unsigned short sA[2][64 * KC];   // 2 x  8 KB
  __shared__ __attribute__((aligned(16))) unsigned short sB[2][128 * KC];  // 2 x 16 KB

  const int tid  = threadIdx.x;
  const int lane = tid & 31;
  const int w    = tid >> 5;       // 0..7
  const int mt   = w & 3;          // 0..3
  const int ng   = w >> 2;         // 0..1
  const int rowBlock = 32 + blockIdx.y * 64;    // H rows (t>=1); last block partial
  const int colBlock = blockIdx.x * 128;
  const int nloc = lane & 15;
  const int half = (lane >> 4) & 1;

  // staging coordinates (per thread)
  const int arow = tid >> 2;                    // 0..63
  const int aseg = (tid & 3) * 16;              // 0,16,32,48 (ushorts)
  int agrow = rowBlock + arow; if (agrow > 4095) agrow = 4095;   // clamp (guarded at store)
  const unsigned short* aSrc = Hbf + (size_t)agrow * G_ + aseg;
  const int bcol = tid >> 1;                    // 0..127
  const int bseg = (tid & 1) * 32;              // 0,32 (ushorts)
  const unsigned short* bSrc = Wout + (size_t)(colBlock + bcol) * G_ + bseg;

  unsigned short* aDst0 = &sA[0][arow * KC + aseg];
  unsigned short* aDst1 = &sA[1][arow * KC + aseg];
  unsigned short* bDst0 = &sB[0][bcol * KC + bseg];
  unsigned short* bDst1 = &sB[1][bcol * KC + bseg];

  // prologue: stage chunk 0
  *(u32x4*)(aDst0)     = *(const u32x4*)(aSrc);
  *(u32x4*)(aDst0 + 8) = *(const u32x4*)(aSrc + 8);
#pragma unroll
  for (int q = 0; q < 4; ++q)
    *(u32x4*)(bDst0 + q * 8) = *(const u32x4*)(bSrc + q * 8);
  __syncthreads();

  v8f acc[4] = {};
  const int rowLoc = mt * 16 + nloc;            // A row within block tile

  for (int kc = 0; kc < G_ / KC; ++kc) {
    const int cur = kc & 1;
    unsigned short* aDstN = (cur ? aDst0 : aDst1);
    unsigned short* bDstN = (cur ? bDst0 : bDst1);
    // stage next chunk while computing current
    if (kc + 1 < G_ / KC) {
      const int k = (kc + 1) * KC;
      __builtin_prefetch(bSrc + k + KC, 0, 1);
      *(u32x4*)(aDstN)     = *(const u32x4*)(aSrc + k);
      *(u32x4*)(aDstN + 8) = *(const u32x4*)(aSrc + k + 8);
#pragma unroll
      for (int q = 0; q < 4; ++q)
        *(u32x4*)(bDstN + q * 8) = *(const u32x4*)(bSrc + k + q * 8);
    }
    // compute on current chunk (fragment loads hit LDS); 2 K-substeps x 4 n-tiles
    const unsigned short* cA = &sA[cur][0];
    const unsigned short* cB = &sB[cur][0];
#pragma unroll
    for (int ks = 0; ks < KC; ks += 32) {
      v16bf a = load_a16(cA, KC, rowLoc, ks, lane);
#pragma unroll
      for (int i = 0; i < 4; ++i) {
        v16bf bb = load_b16(cB, KC, ng * 64 + i * 16 + nloc, ks, lane);
        acc[i] = wmma_bf16(a, bb, acc[i]);
      }
    }
    __syncthreads();
  }

#pragma unroll
  for (int i = 0; i < 4; ++i) {
    int n = colBlock + ng * 64 + i * 16 + nloc;
    float bn = b_out[n];
#pragma unroll
    for (int r = 0; r < 8; ++r) {
      int mg = rowBlock + mt * 16 + r + half * 8;   // H row = t*32 + b, t in [1,127]
      if (mg < 4096) {
        int tt = (mg >> 5) - 1;                     // 0..126
        int bb = mg & 31;
        out[((size_t)bb * 127 + tt) * V_ + n] = acc[i][r] + bn;
      }
    }
  }
}

// ---------------------------------------------------------------------------
// Tail outputs: target_sent (4096) and target_sent_len (32) as floats
// ---------------------------------------------------------------------------
__global__ void tail_copy(const int* __restrict__ sent, const int* __restrict__ slen,
                          float* __restrict__ out) {
  int i = blockIdx.x * blockDim.x + threadIdx.x;
  if (i < B_ * S_) out[LOGITS_ELEMS + i] = (float)sent[i];
  else if (i < B_ * S_ + B_) out[LOGITS_ELEMS + i] = (float)slen[i - B_ * S_];
}

// ---------------------------------------------------------------------------
// Launch
// ---------------------------------------------------------------------------
extern "C" void kernel_launch(void* const* d_in, const int* in_sizes, int n_in,
                              void* d_out, int out_size, void* d_ws, size_t ws_size,
                              hipStream_t stream) {
  const float* entity = (const float*)d_in[0];
  const float* neigh  = (const float*)d_in[1];
  const int*   nnum   = (const int*)d_in[2];
  const float* imgf   = (const float*)d_in[3];
  const int*   sent   = (const int*)d_in[4];
  const int*   slen   = (const int*)d_in[5];
  const float* emb    = (const float*)d_in[6];
  const float* W_ih   = (const float*)d_in[7];
  const float* W_hh   = (const float*)d_in[8];
  const float* b_ih   = (const float*)d_in[9];
  const float* b_hh   = (const float*)d_in[10];
  const float* W_img  = (const float*)d_in[11];
  const float* b_img  = (const float*)d_in[12];
  const float* W_out  = (const float*)d_in[13];
  const float* b_out  = (const float*)d_in[14];
  const float* Wq = (const float*)d_in[15]; const float* bq = (const float*)d_in[16];
  const float* Wk = (const float*)d_in[17]; const float* bk = (const float*)d_in[18];
  const float* Wv = (const float*)d_in[19]; const float* bv = (const float*)d_in[20];
  const float* Wo = (const float*)d_in[21]; const float* bo = (const float*)d_in[22];

  char* ws = (char*)d_ws;
  const size_t oWout = 0;                                         // 65,536,000 B
  const size_t oWhh  = oWout + (size_t)V_ * G_ * 2;               // + 8,388,608
  const size_t oWih  = oWhh  + (size_t)4 * G_ * G_ * 2;           // + 4,194,304
  const size_t oX    = oWih  + (size_t)4 * G_ * E_ * 2;           // + 4,194,304
  const size_t oH    = oX    + (size_t)S_ * B_ * E_ * 2;          // + 8,388,608
  const size_t oC    = oH    + (size_t)S_ * B_ * G_ * 2;          // +   131,072
  const size_t oKV   = oC    + (size_t)B_ * G_ * 4;               // + 2,097,152

  unsigned short* Woutb = (unsigned short*)(ws + oWout);
  unsigned short* Whhb  = (unsigned short*)(ws + oWhh);
  unsigned short* Wihb  = (unsigned short*)(ws + oWih);
  unsigned short* Xbf   = (unsigned short*)(ws + oX);
  unsigned short* Hbf   = (unsigned short*)(ws + oH);
  float*          Ccar  = (float*)(ws + oC);
  float*          kbuf  = (float*)(ws + oKV);
  float*          vbuf  = kbuf + (size_t)B_ * NN_ * HI_;

  // One-time bf16 conversions (L2-resident weights; GEMMs run on WMMA bf16 pipe)
  {
    int n1 = V_ * G_;       cvt_bf16<<<(n1 + 255) / 256, 256, 0, stream>>>(W_out, Woutb, n1);
    int n2 = 4 * G_ * G_;   cvt_bf16<<<(n2 + 255) / 256, 256, 0, stream>>>(W_hh, Whhb, n2);
    int n3 = 4 * G_ * E_;   cvt_bf16<<<(n3 + 255) / 256, 256, 0, stream>>>(W_ih, Wihb, n3);
  }

  // MHA + img_embed -> X row t=0
  attn_kernel<<<B_, 256, 0, stream>>>(entity, neigh, nnum, imgf,
                                      Wq, bq, Wk, bk, Wv, bv, Wo, bo,
                                      W_img, b_img, kbuf, vbuf, Xbf);
  // Embedding gather -> X rows t=1..127
  gather_embed<<<(S_ - 1) * B_, 256, 0, stream>>>(sent, emb, Xbf);

  // Serial LSTM recurrence (h feedback forces per-step launches)
  for (int t = 0; t < S_; ++t)
    lstm_step_kernel<<<32, 128, 0, stream>>>(Xbf, Hbf, Wihb, Whhb, b_ih, b_hh, Ccar, t);

  // Dominant GEMM: 4064 x 32000 x 1024 bf16 WMMA, f32 accumulate, LDS-tiled
  out_gemm<<<dim3(V_ / 128, 64), 256, 0, stream>>>(Hbf, Woutb, b_out, (float*)d_out);

  // Passthrough int outputs
  tail_copy<<<(B_ * S_ + B_ + 255) / 256, 256, 0, stream>>>(sent, slen, (float*)d_out);
}
